// TopicAdaptiveLSTMCell_88115549045316
// MI455X (gfx1250) — matrix operand order repaired and checked by reading, not compiled, not run on hardware
//
#include <hip/hip_runtime.h>

#define B_    4096
#define IN_   1024
#define H_    1024
#define F_    128
#define T_    3
#define G4H   4096
#define LN_EPS 1e-5f

typedef __bf16 bf16;
typedef bf16  v16bf __attribute__((ext_vector_type(16)));
typedef bf16  v8bf  __attribute__((ext_vector_type(8)));
typedef bf16  v4bf  __attribute__((ext_vector_type(4)));
typedef float v8f   __attribute__((ext_vector_type(8)));

// ---------------------------------------------------------------------------
// Kernel 0: streaming fp32 -> bf16 conversion (4 elems / thread).
// Run once for input_, hx, wic, whc, wia, wha so the GEMM hot loops are
// conversion-free and read half the bytes.
// ---------------------------------------------------------------------------
__global__ __launch_bounds__(256) void ta_cvt_kernel(
    const float* __restrict__ src, bf16* __restrict__ dst) {
  int i = blockIdx.x * blockDim.x + threadIdx.x;
  float4 v = ((const float4*)src)[i];
  v4bf o;
  o[0] = (bf16)v.x; o[1] = (bf16)v.y; o[2] = (bf16)v.z; o[3] = (bf16)v.w;
  ((v4bf*)dst)[i] = o;
}

// ---------------------------------------------------------------------------
// Kernel 1: topic -> per-(b,f) scales for the i- and h-paths.  Tiny (T=3).
// ---------------------------------------------------------------------------
__global__ __launch_bounds__(256) void ta_scales_kernel(
    const float* __restrict__ topic,
    const float* __restrict__ th_iw, const float* __restrict__ th_ib,
    const float* __restrict__ th_hw, const float* __restrict__ th_hb,
    const float* __restrict__ wib,   const float* __restrict__ whb,
    float* __restrict__ Si, float* __restrict__ Sh) {
  int idx = blockIdx.x * blockDim.x + threadIdx.x;   // over B*F
  int b = idx >> 7;
  int f = idx & (F_ - 1);
  float tw[T_];
#pragma unroll
  for (int t = 0; t < T_; ++t) tw[t] = topic[b * T_ + t];
  float si = 0.f, sh = 0.f;
#pragma unroll
  for (int t = 0; t < T_; ++t) {
    float thi = th_ib[t], thh = th_hb[t];
#pragma unroll
    for (int u = 0; u < T_; ++u) {
      thi += th_iw[t * T_ + u] * tw[u];
      thh += th_hw[t * T_ + u] * tw[u];
    }
    si += thi * wib[f * T_ + t];
    sh += thh * whb[f * T_ + t];
  }
  Si[idx] = si;
  Sh[idx] = sh;
}

// ---------------------------------------------------------------------------
// Kernel 2: U = (X @ Wc^T) * S  -> bf16 [B,128].  WMMA bf16, K=1024.
// A-tile (16 x 1024 bf16 = 32 KB, contiguous) is async-DMA'd to LDS once per
// workgroup; all 8 waves build A fragments from LDS.  blockIdx.z selects path.
// ---------------------------------------------------------------------------
__global__ __launch_bounds__(256) void ta_gemm1_kernel(
    const bf16* __restrict__ Xi, const bf16* __restrict__ Xh,
    const bf16* __restrict__ Wic, const bf16* __restrict__ Whc,
    const float* __restrict__ Si, const float* __restrict__ Sh,
    bf16* __restrict__ Ui, bf16* __restrict__ Uh) {
  __shared__ bf16 xtile[16 * IN_];              // 32 KB
  const int path = blockIdx.z;
  const bf16*  X = path ? Xh  : Xi;
  const bf16*  W = path ? Whc : Wic;
  const float* S = path ? Sh  : Si;
  bf16*        U = path ? Uh  : Ui;

  const int lane = threadIdx.x & 31;
  const int wv   = threadIdx.x >> 5;
  const int m0   = blockIdx.x * 16;
  const int n0   = wv * 16;
  const int hi   = lane >> 4;
  const int lr   = lane & 15;

  // Async copy: 256 threads x 16 B x 8 = 32 KB (rows m0..m0+15 are contiguous).
  {
    unsigned    lbase = (unsigned)(uintptr_t)(void*)xtile + threadIdx.x * 16u;
    const char* gbase = (const char*)(X + (size_t)m0 * IN_) + threadIdx.x * 16u;
#pragma unroll
    for (int i = 0; i < 8; ++i) {
      unsigned    l = lbase + i * 4096u;
      const char* g = gbase + i * 4096u;
      asm volatile("global_load_async_to_lds_b128 %0, %1, off"
                   :: "v"(l), "v"(g) : "memory");
    }
    asm volatile("s_wait_asynccnt 0" ::: "memory");
  }
  __syncthreads();

  const bf16* arow = xtile + lr * IN_ + hi * 8;                 // LDS
  const bf16* Wrow = W + (size_t)(n0 + lr) * IN_ + hi * 8;      // global bf16

  v8f acc = {};
  for (int k0 = 0; k0 < IN_; k0 += 32) {
    v8bf alo = *(const v8bf*)(arow + k0);
    v8bf ahi = *(const v8bf*)(arow + k0 + 16);
    v16bf a = __builtin_shufflevector(alo, ahi,
                                      0,1,2,3,4,5,6,7,8,9,10,11,12,13,14,15);
    v8bf blo = *(const v8bf*)(Wrow + k0);
    v8bf bhi = *(const v8bf*)(Wrow + k0 + 16);
    v16bf b = __builtin_shufflevector(blo, bhi,
                                      0,1,2,3,4,5,6,7,8,9,10,11,12,13,14,15);
    acc = __builtin_amdgcn_wmma_f32_16x16x32_bf16(false, a, false, b,
                                                  (short)0, acc, false, false);
  }

  const int ncol = n0 + lr;
#pragma unroll
  for (int v = 0; v < 8; ++v) {
    int m = m0 + v + 8 * hi;
    float out = acc[v] * S[(size_t)m * F_ + ncol];
    U[(size_t)m * F_ + ncol] = (bf16)out;
  }
}

// ---------------------------------------------------------------------------
// Kernel 3: ig = Ui @ Wia^T, hg = Uh @ Wha^T (bf16 out, [B,4096]) with fused
// per-row LN statistics.  Workgroup = 16 rows x all 4096 cols, both matrices.
// Ui/Uh tiles (4 KB each, contiguous) async-DMA'd to LDS; A fragments built
// from LDS once and reused for all 256 column tiles.
// ---------------------------------------------------------------------------
__global__ __launch_bounds__(256) void ta_gemm2_kernel(
    const bf16* __restrict__ Ui, const bf16* __restrict__ Uh,
    const bf16* __restrict__ Wia, const bf16* __restrict__ Wha,
    bf16* __restrict__ ig, bf16* __restrict__ hg,
    float* __restrict__ stats) {
  __shared__ bf16  uitile[16 * F_];   // 4 KB
  __shared__ bf16  uhtile[16 * F_];   // 4 KB
  __shared__ float sred[16 * 4];      // per-row: sum_i, ss_i, sum_h, ss_h

  const int lane = threadIdx.x & 31;
  const int wv   = threadIdx.x >> 5;
  const int m0   = blockIdx.x * 16;
  const int hi   = lane >> 4;
  const int lr   = lane & 15;

  {
    unsigned    li = (unsigned)(uintptr_t)(void*)uitile + threadIdx.x * 16u;
    unsigned    lh = (unsigned)(uintptr_t)(void*)uhtile + threadIdx.x * 16u;
    const char* gi = (const char*)(Ui + (size_t)m0 * F_) + threadIdx.x * 16u;
    const char* gh = (const char*)(Uh + (size_t)m0 * F_) + threadIdx.x * 16u;
    asm volatile("global_load_async_to_lds_b128 %0, %1, off"
                 :: "v"(li), "v"(gi) : "memory");
    asm volatile("global_load_async_to_lds_b128 %0, %1, off"
                 :: "v"(lh), "v"(gh) : "memory");
    asm volatile("s_wait_asynccnt 0" ::: "memory");
  }
  if (threadIdx.x < 64) sred[threadIdx.x] = 0.f;
  __syncthreads();

  // Build per-wave A fragments (K = 128 -> 4 operands per matrix) from LDS.
  const bf16* uir = uitile + lr * F_ + hi * 8;
  const bf16* uhr = uhtile + lr * F_ + hi * 8;
  v16bf ai[4], ah[4];
#pragma unroll
  for (int q = 0; q < 4; ++q) {
    v8bf l0 = *(const v8bf*)(uir + q * 32);
    v8bf h0 = *(const v8bf*)(uir + q * 32 + 16);
    ai[q] = __builtin_shufflevector(l0, h0,
                                    0,1,2,3,4,5,6,7,8,9,10,11,12,13,14,15);
    v8bf l1 = *(const v8bf*)(uhr + q * 32);
    v8bf h1 = *(const v8bf*)(uhr + q * 32 + 16);
    ah[q] = __builtin_shufflevector(l1, h1,
                                    0,1,2,3,4,5,6,7,8,9,10,11,12,13,14,15);
  }

  float sum_i[8] = {}, ss_i[8] = {}, sum_h[8] = {}, ss_h[8] = {};

  for (int t = wv; t < 256; t += 8) {          // 16-col tiles of N=4096
    const int ncol = t * 16 + lr;
    const bf16* Wiarow = Wia + (size_t)ncol * F_ + hi * 8;
    const bf16* Wharow = Wha + (size_t)ncol * F_ + hi * 8;
    if (t + 8 < 256) {                         // prefetch next weight tile
      __builtin_prefetch(Wia + (size_t)(ncol + 128) * F_, 0, 0);
      __builtin_prefetch(Wha + (size_t)(ncol + 128) * F_, 0, 0);
    }
    v8f ci = {}, ch = {};
#pragma unroll
    for (int q = 0; q < 4; ++q) {
      v16bf b = __builtin_shufflevector(*(const v8bf*)(Wiarow + q * 32),
                                        *(const v8bf*)(Wiarow + q * 32 + 16),
                                        0,1,2,3,4,5,6,7,8,9,10,11,12,13,14,15);
      ci = __builtin_amdgcn_wmma_f32_16x16x32_bf16(false, ai[q], false, b,
                                                   (short)0, ci, false, false);
    }
#pragma unroll
    for (int q = 0; q < 4; ++q) {
      v16bf b = __builtin_shufflevector(*(const v8bf*)(Wharow + q * 32),
                                        *(const v8bf*)(Wharow + q * 32 + 16),
                                        0,1,2,3,4,5,6,7,8,9,10,11,12,13,14,15);
      ch = __builtin_amdgcn_wmma_f32_16x16x32_bf16(false, ah[q], false, b,
                                                   (short)0, ch, false, false);
    }
#pragma unroll
    for (int v = 0; v < 8; ++v) {
      int m = m0 + v + 8 * hi;
      float vi = ci[v], vh = ch[v];
      sum_i[v] += vi; ss_i[v] += vi * vi;
      sum_h[v] += vh; ss_h[v] += vh * vh;
      ig[(size_t)m * G4H + ncol] = (bf16)vi;
      hg[(size_t)m * G4H + ncol] = (bf16)vh;
    }
  }

#pragma unroll
  for (int v = 0; v < 8; ++v) {
    int r = v + 8 * hi;
    atomicAdd(&sred[r * 4 + 0], sum_i[v]);
    atomicAdd(&sred[r * 4 + 1], ss_i[v]);
    atomicAdd(&sred[r * 4 + 2], sum_h[v]);
    atomicAdd(&sred[r * 4 + 3], ss_h[v]);
  }
  __syncthreads();

  if (threadIdx.x < 16) {
    int r = threadIdx.x;
    float inv = 1.0f / (float)G4H;
    float mu_i  = sred[r * 4 + 0] * inv;
    float var_i = sred[r * 4 + 1] * inv - mu_i * mu_i;
    float mu_h  = sred[r * 4 + 2] * inv;
    float var_h = sred[r * 4 + 3] * inv - mu_h * mu_h;
    float* st = stats + (size_t)(m0 + r) * 4;
    st[0] = mu_i; st[1] = rsqrtf(var_i + LN_EPS);
    st[2] = mu_h; st[3] = rsqrtf(var_h + LN_EPS);
  }
}

// ---------------------------------------------------------------------------
// Kernel 4: LN(ig)+LN(hg) -> gates -> c_pre -> LN over H -> cy, hy.
// One workgroup per row b; 256 threads x 4 cols.
// ---------------------------------------------------------------------------
__global__ __launch_bounds__(256) void ta_lstm_kernel(
    const bf16* __restrict__ ig, const bf16* __restrict__ hg,
    const float* __restrict__ stats, const float* __restrict__ cx,
    const float* __restrict__ lniw, const float* __restrict__ lnib,
    const float* __restrict__ lnhw, const float* __restrict__ lnhb,
    const float* __restrict__ lncw, const float* __restrict__ lncb,
    float* __restrict__ out) {
  __shared__ float ssum[256], sss[256], smu[2];
  const int b   = blockIdx.x;
  const int tid = threadIdx.x;
  const float mu_i = stats[b * 4 + 0], rs_i = stats[b * 4 + 1];
  const float mu_h = stats[b * 4 + 2], rs_h = stats[b * 4 + 3];
  const bf16* igr = ig + (size_t)b * G4H;
  const bf16* hgr = hg + (size_t)b * G4H;

  float cpre[4], og[4];
  float lsum = 0.f, lss = 0.f;
#pragma unroll
  for (int u = 0; u < 4; ++u) {
    int j = tid + 256 * u;
    float gi = ((float)igr[j]          - mu_i) * rs_i * lniw[j]          + lnib[j]
             + ((float)hgr[j]          - mu_h) * rs_h * lnhw[j]          + lnhb[j];
    float gf = ((float)igr[H_ + j]     - mu_i) * rs_i * lniw[H_ + j]     + lnib[H_ + j]
             + ((float)hgr[H_ + j]     - mu_h) * rs_h * lnhw[H_ + j]     + lnhb[H_ + j];
    float gg = ((float)igr[2 * H_ + j] - mu_i) * rs_i * lniw[2 * H_ + j] + lnib[2 * H_ + j]
             + ((float)hgr[2 * H_ + j] - mu_h) * rs_h * lnhw[2 * H_ + j] + lnhb[2 * H_ + j];
    float go = ((float)igr[3 * H_ + j] - mu_i) * rs_i * lniw[3 * H_ + j] + lnib[3 * H_ + j]
             + ((float)hgr[3 * H_ + j] - mu_h) * rs_h * lnhw[3 * H_ + j] + lnhb[3 * H_ + j];
    float iv = 1.f / (1.f + __expf(-gi));
    float fv = 1.f / (1.f + __expf(-gf));
    float gv = tanhf(gg);
    float ov = 1.f / (1.f + __expf(-go));
    float c  = fv * cx[(size_t)b * H_ + j] + iv * gv;
    cpre[u] = c; og[u] = ov;
    lsum += c; lss += c * c;
  }
  ssum[tid] = lsum; sss[tid] = lss;
  __syncthreads();
  for (int s = 128; s > 0; s >>= 1) {
    if (tid < s) { ssum[tid] += ssum[tid + s]; sss[tid] += sss[tid + s]; }
    __syncthreads();
  }
  if (tid == 0) {
    float mu  = ssum[0] / (float)H_;
    float var = sss[0] / (float)H_ - mu * mu;
    smu[0] = mu; smu[1] = rsqrtf(var + LN_EPS);
  }
  __syncthreads();
  const float mu_c = smu[0], rs_c = smu[1];
#pragma unroll
  for (int u = 0; u < 4; ++u) {
    int j = tid + 256 * u;
    float cy = (cpre[u] - mu_c) * rs_c * lncw[j] + lncb[j];
    float hy = og[u] * tanhf(cy);
    out[(size_t)b * H_ + j]                   = hy;
    out[(size_t)B_ * H_ + (size_t)b * H_ + j] = cy;
  }
}

// ---------------------------------------------------------------------------
extern "C" void kernel_launch(void* const* d_in, const int* in_sizes, int n_in,
                              void* d_out, int out_size, void* d_ws, size_t ws_size,
                              hipStream_t stream) {
  (void)in_sizes; (void)n_in; (void)out_size; (void)ws_size;
  const float* input_ = (const float*)d_in[0];
  const float* hx     = (const float*)d_in[1];
  const float* cx     = (const float*)d_in[2];
  const float* topic  = (const float*)d_in[3];
  const float* wia    = (const float*)d_in[4];
  const float* wib    = (const float*)d_in[5];
  const float* wic    = (const float*)d_in[6];
  const float* wha    = (const float*)d_in[7];
  const float* whb    = (const float*)d_in[8];
  const float* whc    = (const float*)d_in[9];
  const float* thiw   = (const float*)d_in[10];
  const float* thib   = (const float*)d_in[11];
  const float* thhw   = (const float*)d_in[12];
  const float* thhb   = (const float*)d_in[13];
  const float* lniw   = (const float*)d_in[14];
  const float* lnib   = (const float*)d_in[15];
  const float* lnhw   = (const float*)d_in[16];
  const float* lnhb   = (const float*)d_in[17];
  const float* lncw   = (const float*)d_in[18];
  const float* lncb   = (const float*)d_in[19];
  float* out = (float*)d_out;

  // workspace layout (MB offsets)
  char* ws = (char*)d_ws;
  float* Si    = (float*)(ws + 0);                       // 2 MB
  float* Sh    = (float*)(ws + (2ull  << 20));           // 2 MB
  bf16*  Ui    = (bf16*) (ws + (4ull  << 20));           // 1 MB
  bf16*  Uh    = (bf16*) (ws + (5ull  << 20));           // 1 MB
  bf16*  igb   = (bf16*) (ws + (6ull  << 20));           // 32 MB
  bf16*  hgb   = (bf16*) (ws + (38ull << 20));           // 32 MB
  float* stat  = (float*)(ws + (70ull << 20));           // 64 KB
  bf16*  Xib   = (bf16*) (ws + (71ull << 20));           // 8 MB
  bf16*  Xhb   = (bf16*) (ws + (79ull << 20));           // 8 MB
  bf16*  Wicb  = (bf16*) (ws + (87ull << 20));                   // 256 KB
  bf16*  Whcb  = (bf16*) (ws + (87ull << 20) + (256u << 10));    // 256 KB
  bf16*  Wiab  = (bf16*) (ws + (87ull << 20) + (512u << 10));    // 1 MB
  bf16*  Whab  = (bf16*) (ws + (87ull << 20) + (1536u << 10));   // 1 MB

  // fp32 -> bf16 pre-conversion (activations + reused weights)
  ta_cvt_kernel<<<dim3((B_ * IN_)  / 1024), 256, 0, stream>>>(input_, Xib);
  ta_cvt_kernel<<<dim3((B_ * H_)   / 1024), 256, 0, stream>>>(hx,     Xhb);
  ta_cvt_kernel<<<dim3((F_ * IN_)  / 1024), 256, 0, stream>>>(wic,    Wicb);
  ta_cvt_kernel<<<dim3((F_ * H_)   / 1024), 256, 0, stream>>>(whc,    Whcb);
  ta_cvt_kernel<<<dim3((G4H * F_)  / 1024), 256, 0, stream>>>(wia,    Wiab);
  ta_cvt_kernel<<<dim3((G4H * F_)  / 1024), 256, 0, stream>>>(wha,    Whab);

  ta_scales_kernel<<<dim3((B_ * F_) / 256), 256, 0, stream>>>(
      topic, thiw, thib, thhw, thhb, wib, whb, Si, Sh);

  ta_gemm1_kernel<<<dim3(B_ / 16, 1, 2), 256, 0, stream>>>(
      Xib, Xhb, Wicb, Whcb, Si, Sh, Ui, Uh);

  ta_gemm2_kernel<<<dim3(B_ / 16), 256, 0, stream>>>(
      Ui, Uh, Wiab, Whab, igb, hgb, stat);

  ta_lstm_kernel<<<dim3(B_), 256, 0, stream>>>(
      igb, hgb, stat, cx, lniw, lnib, lnhw, lnhb, lncw, lncb, out);
}
